// SwinTransformerBlock_13408887898421
// MI455X (gfx1250) — matrix operand order repaired and checked by reading, not compiled, not run on hardware
//
#include <hip/hip_runtime.h>

typedef _Float16 half_t;
typedef __attribute__((ext_vector_type(16))) _Float16 v16h;
typedef __attribute__((ext_vector_type(8)))  _Float16 v8h;
typedef __attribute__((ext_vector_type(4)))  _Float16 v4h;
typedef __attribute__((ext_vector_type(8)))  float    v8f;

#define D_    128
#define NTOK  131072   // B*N = 8*16384
#define NWTOT 8192     // total windows = 8*1024

// ---------------- WMMA helpers ----------------

__device__ __forceinline__ v8f wmma_f16(v16h a, v16h b, v8f c) {
  // D = A(16x32 f16) * B(32x16 f16) + C(16x16 f32)
  return __builtin_amdgcn_wmma_f32_16x16x32_f16(false, a, false, b, (short)0, c, false, false);
}

// A-fragment (16x32 f16): per lane one contiguous 16-half run:
//   elements 0..15 -> K = kb + (lane>=16?16:0) + e
// M = lane & 15. Source row-major [16 x ld].
__device__ __forceinline__ v16h load_a_frag(const half_t* __restrict__ base, int ld,
                                            int kb, int lane) {
  const half_t* p = base + (size_t)(lane & 15) * ld + kb + ((lane >> 4) << 4);
  return *(const v16h*)p;
}

// B-fragment (32x16 f16), same contiguous k order as A: lanes 0-15 hold K=kb+0..15 for
// column n=lane, lanes 16-31 hold K=kb+16..31 for n=lane-16 -> ONE contiguous
// 16-half run per lane. Source is row-major [N x K] (weights W[n,k], or any
// [row=n][col=k] buffer).
__device__ __forceinline__ v16h load_b_frag(const half_t* __restrict__ base, int ld,
                                            int nb, int kb, int lane) {
  const half_t* p = base + (size_t)(nb + (lane & 15)) * ld + kb + ((lane >> 4) << 4);
  return *(const v16h*)p;
}

// ---------------- scalar helpers ----------------

__device__ __forceinline__ float gelu_exact(float x) {
  return 0.5f * x * (1.0f + erff(x * 0.70710678118654752f));
}

__device__ __forceinline__ float wave_sum(float v) {
#pragma unroll
  for (int m = 16; m; m >>= 1) v += __shfl_xor(v, m, 32);
  return v;
}

// LayerNorm of one 128-elem row spread 4-per-lane over a wave32.
__device__ __forceinline__ float4 ln_row(float4 xv, const float* __restrict__ g,
                                         const float* __restrict__ b, int lane) {
  float mean = wave_sum(xv.x + xv.y + xv.z + xv.w) * (1.0f / 128.0f);
  float dx = xv.x - mean, dy = xv.y - mean, dz = xv.z - mean, dw = xv.w - mean;
  float var = wave_sum(dx * dx + dy * dy + dz * dz + dw * dw) * (1.0f / 128.0f);
  float rs = rsqrtf(var + 1e-5f);
  float4 gv = ((const float4*)g)[lane];
  float4 bv = ((const float4*)b)[lane];
  float4 o;
  o.x = dx * rs * gv.x + bv.x;
  o.y = dy * rs * gv.y + bv.y;
  o.z = dz * rs * gv.z + bv.z;
  o.w = dw * rs * gv.w + bv.w;
  return o;
}

__device__ __forceinline__ void store_h4(half_t* dst, float4 y) {
  v4h h;
  h[0] = (half_t)y.x; h[1] = (half_t)y.y; h[2] = (half_t)y.z; h[3] = (half_t)y.w;
  *(v4h*)dst = h;
}

// ---------------- kernels ----------------

__global__ void cast_w_kernel(const float* __restrict__ src, half_t* __restrict__ dst, int n) {
  int i = blockIdx.x * blockDim.x + threadIdx.x;
  if (i < n) dst[i] = (half_t)src[i];
}

// LN(ln1) of x[B,N,D] and scatter into window-gathered order [B,nW,16,D] as f16.
__global__ void ln_gather1_kernel(const float* __restrict__ X, const float* __restrict__ g,
                                  const float* __restrict__ b, half_t* __restrict__ O) {
  int gid = blockIdx.x * blockDim.x + threadIdx.x;
  int t = gid >> 5, lane = gid & 31;
  float4 xv = ((const float4*)(X + (size_t)t * D_))[lane];
  float4 y = ln_row(xv, g, b, lane);
  int bi = t >> 14, n = t & 16383;
  int r = n >> 7, c = n & 127;
  int w = (r >> 2) * 32 + (c >> 2);
  int p = ((r & 3) << 2) + (c & 3);
  size_t dst = ((size_t)((bi << 10) + w) * 16 + p) * D_ + lane * 4;
  store_h4(O + dst, y);
}

// Branch-2 pre-attn: source is out1 in window-order; fuse win->img, roll(-2,-2),
// img->win into the source index; LN(ln3); emit f16 (dest linear window order).
__global__ void ln_roll_gather_kernel(const float* __restrict__ X, const float* __restrict__ g,
                                      const float* __restrict__ b, half_t* __restrict__ O) {
  int gid = blockIdx.x * blockDim.x + threadIdx.x;
  int t = gid >> 5, lane = gid & 31;
  int bi = t >> 14, n = t & 16383;
  int w2 = n >> 4, p2 = n & 15;
  int r2 = (w2 >> 5) * 4 + (p2 >> 2);
  int c2 = (w2 & 31) * 4 + (p2 & 3);
  int rs = (r2 + 2) & 127, cs = (c2 + 2) & 127;
  int st = ((rs >> 2) * 32 + (cs >> 2)) * 16 + ((rs & 3) << 2) + (cs & 3);
  float4 xv = ((const float4*)(X + ((size_t)(bi << 14) + st) * D_))[lane];
  float4 y = ln_row(xv, g, b, lane);
  store_h4(O + (size_t)t * D_ + lane * 4, y);
}

// Plain LN f32 -> f16 (for MLP input).
__global__ void ln_plain_kernel(const float* __restrict__ X, const float* __restrict__ g,
                                const float* __restrict__ b, half_t* __restrict__ O) {
  int gid = blockIdx.x * blockDim.x + threadIdx.x;
  int t = gid >> 5, lane = gid & 31;
  float4 xv = ((const float4*)(X + (size_t)t * D_))[lane];
  float4 y = ln_row(xv, g, b, lane);
  store_h4(O + (size_t)t * D_ + lane * 4, y);
}

// res1 = trunc(x) + attn1 (both linear order).
__global__ void res1_kernel(const float* __restrict__ X, const float* __restrict__ A,
                            float* __restrict__ R) {
  size_t i = (size_t)blockIdx.x * blockDim.x + threadIdx.x;
  float4 x = ((const float4*)X)[i];
  float4 a = ((const float4*)A)[i];
  float4 o;
  o.x = truncf(x.x) + a.x; o.y = truncf(x.y) + a.y;
  o.z = truncf(x.z) + a.z; o.w = truncf(x.w) + a.w;
  ((float4*)R)[i] = o;
}

// res2 = out1 + unroll(+2,+2) of attn2 (window-order both sides).
__global__ void res2_kernel(const float* __restrict__ O1, const float* __restrict__ A2,
                            float* __restrict__ R) {
  int gid = blockIdx.x * blockDim.x + threadIdx.x;
  int t = gid >> 5, lane = gid & 31;
  int bi = t >> 14, n = t & 16383;
  int w = n >> 4, p = n & 15;
  int r = (w >> 5) * 4 + (p >> 2);
  int c = (w & 31) * 4 + (p & 3);
  int rs = (r - 2) & 127, cs = (c - 2) & 127;
  int st = ((rs >> 2) * 32 + (cs >> 2)) * 16 + ((rs & 3) << 2) + (cs & 3);
  float4 a  = ((const float4*)(A2 + ((size_t)(bi << 14) + st) * D_))[lane];
  float4 o1 = ((const float4*)(O1 + (size_t)t * D_))[lane];
  float4 o;
  o.x = o1.x + a.x; o.y = o1.y + a.y; o.z = o1.z + a.z; o.w = o1.w + a.w;
  ((float4*)(R + (size_t)t * D_))[lane] = o;
}

// Full windowed MHSA: one wave per 16-token window. 2 waves / 64-thread block.
__global__ void attn_kernel(const half_t* __restrict__ Xw,
                            const half_t* __restrict__ Wqkv, const float* __restrict__ bqkv,
                            const half_t* __restrict__ Wo,   const float* __restrict__ bo,
                            float* __restrict__ Y) {
  __shared__ __align__(32) half_t sQK[2][16][256];   // Q cols 0..127, K cols 128..255
  __shared__ __align__(32) half_t sV[2][128][32];    // V^T [d][token], token 16..31 = 0 pad
  __shared__ __align__(32) half_t sP[2][16][32];     // softmax probs, K-pad 16..31 = 0
  __shared__ __align__(32) half_t sO[2][16][128];    // attention output
  int lane = threadIdx.x & 31;
  int wv   = threadIdx.x >> 5;
  int win  = blockIdx.x * 2 + wv;
  int nlo  = lane & 15;
  int mo   = (lane >> 4) << 3;
  const half_t* xa = Xw + (size_t)win * (16 * D_);

  // zero V K-padding (token columns 16..31)
  for (int i = lane; i < 128 * 16; i += 32)
    sV[wv][i >> 4][16 + (i & 15)] = (half_t)0.0f;

  // ---- QKV projection: [16,128] @ [128,384]^T ----
  for (int nt = 0; nt < 16; ++nt) {           // Q and K
    v8f c = {};
#pragma unroll
    for (int kb = 0; kb < 128; kb += 32)
      c = wmma_f16(load_a_frag(xa, 128, kb, lane),
                   load_b_frag(Wqkv, 128, nt * 16, kb, lane), c);
    int n = nt * 16 + nlo;
    float bv = bqkv[n];
#pragma unroll
    for (int r = 0; r < 8; ++r) sQK[wv][mo + r][n] = (half_t)(c[r] + bv);
  }
  for (int nt = 16; nt < 24; ++nt) {          // V, stored transposed
    v8f c = {};
#pragma unroll
    for (int kb = 0; kb < 128; kb += 32)
      c = wmma_f16(load_a_frag(xa, 128, kb, lane),
                   load_b_frag(Wqkv, 128, nt * 16, kb, lane), c);
    int n = nt * 16 + nlo;
    float bv = bqkv[n];
    int cv = n - 256;
#pragma unroll
    for (int r = 0; r < 8; ++r) sV[wv][cv][mo + r] = (half_t)(c[r] + bv);
  }
  __syncthreads();

  // ---- per-head attention ----
  for (int h = 0; h < 2; ++h) {
    const half_t* qb = &sQK[wv][0][h * 64];
    const half_t* kk = &sQK[wv][0][128 + h * 64];
    const half_t* vb = &sV[wv][h * 64][0];
    v8f s = {};
    s = wmma_f16(load_a_frag(qb, 256, 0,  lane), load_b_frag(kk, 256, 0, 0,  lane), s);
    s = wmma_f16(load_a_frag(qb, 256, 32, lane), load_b_frag(kk, 256, 0, 32, lane), s);
    // softmax over keys = across lanes 0..15 / 16..31 per row
#pragma unroll
    for (int r = 0; r < 8; ++r) {
      float v = s[r] * 0.125f;          // 1/sqrt(64)
      float m = v;
#pragma unroll
      for (int msk = 8; msk; msk >>= 1) m = fmaxf(m, __shfl_xor(m, msk, 32));
      float e = __expf(v - m);
      float sum = e;
#pragma unroll
      for (int msk = 8; msk; msk >>= 1) sum += __shfl_xor(sum, msk, 32);
      sP[wv][mo + r][nlo] = (half_t)(e / sum);
      sP[wv][mo + r][16 + nlo] = (half_t)0.0f;
    }
    __syncthreads();
    // O = P @ V  (K padded 16->32 with zeros; lanes>=16 of B read the zero pad)
    v16h pa = load_a_frag(&sP[wv][0][0], 32, 0, lane);
#pragma unroll
    for (int nt = 0; nt < 4; ++nt) {
      v8f o = {};
      o = wmma_f16(pa, load_b_frag(vb, 32, nt * 16, 0, lane), o);
      int oc = h * 64 + nt * 16 + nlo;
#pragma unroll
      for (int r = 0; r < 8; ++r) sO[wv][mo + r][oc] = (half_t)o[r];
    }
    __syncthreads();
  }

  // ---- output projection: [16,128] @ [128,128]^T ----
  float* yb = Y + (size_t)win * (16 * D_);
  for (int nt = 0; nt < 8; ++nt) {
    v8f c = {};
#pragma unroll
    for (int kb = 0; kb < 128; kb += 32)
      c = wmma_f16(load_a_frag(&sO[wv][0][0], 128, kb, lane),
                   load_b_frag(Wo, 128, nt * 16, kb, lane), c);
    int n = nt * 16 + nlo;
    float bv = bo[n];
#pragma unroll
    for (int r = 0; r < 8; ++r) yb[(size_t)(mo + r) * D_ + n] = c[r] + bv;
  }
}

// Y = gelu(A @ W^T + bias)  [+ resid -> f32, else -> f16].
// Block (8 waves) stages an A row-panel in LDS via the CDNA5 async global->LDS
// engine (GLOBAL_LOAD_ASYNC_TO_LDS_B128, ASYNCcnt), then each wave computes a
// 16x64 tile (4 accumulators, A-frag reused 4x). Requires N % 64 == 0 and
// (N/64) dividing 8. Dynamic LDS = (8/(N/64)) * 16 * K * 2 bytes.
__global__ void gemm_kernel(const half_t* __restrict__ A, const half_t* __restrict__ W,
                            const float* __restrict__ bias, const float* __restrict__ resid,
                            half_t* __restrict__ Yh, float* __restrict__ Yf,
                            int M, int N, int K) {
  extern __shared__ __align__(32) half_t sA[];
  int tid = threadIdx.x;
  int lane = tid & 31, wave = tid >> 5;
  int ngroups = N >> 6;              // #64-col groups (8 for N=512, 2 for N=128)
  int mtPerBlock = 8 / ngroups;      // 1 (N=512) or 4 (N=128)
  int mt0 = blockIdx.x * mtPerBlock;

  // async-stage A rows [mt0*16, (mt0+mtPerBlock)*16) into LDS: 16B per lane per op.
  // LDS generic pointers carry the LDS byte offset in their low 32 bits.
  {
    int totalBytes = mtPerBlock * 16 * K * 2;
    unsigned ldsBase = (unsigned)(unsigned long long)(&sA[0]);
    const char* gBase = (const char*)(A + (size_t)mt0 * 16 * K);
    for (int off = tid * 16; off < totalBytes; off += 256 * 16) {
      unsigned l = ldsBase + off;
      const char* g = gBase + off;
      asm volatile("global_load_async_to_lds_b128 %0, %1, off"
                   :: "v"(l), "v"(g)
                   : "memory");
    }
    asm volatile("s_wait_asynccnt 0x0" ::: "memory");
  }
  __syncthreads();

  int mtl = wave / ngroups;          // this wave's local mt
  int ng  = wave % ngroups;          // this wave's 64-col group
  const half_t* abase = sA + (size_t)mtl * 16 * K;
  v8f acc[4] = {};
  for (int kb = 0; kb < K; kb += 32) {
    v16h af = load_a_frag(abase, K, kb, lane);
#pragma unroll
    for (int j = 0; j < 4; ++j)
      acc[j] = wmma_f16(af, load_b_frag(W, K, (ng * 4 + j) * 16, kb, lane), acc[j]);
  }
  int mo = (lane >> 4) << 3;
  int mrow0 = (mt0 + mtl) * 16 + mo;
#pragma unroll
  for (int j = 0; j < 4; ++j) {
    int n = (ng * 4 + j) * 16 + (lane & 15);
    float bv = bias[n];
#pragma unroll
    for (int r = 0; r < 8; ++r) {
      size_t idx = (size_t)(mrow0 + r) * N + n;
      float gy = gelu_exact(acc[j][r] + bv);
      if (Yf) Yf[idx] = gy + resid[idx];
      else    Yh[idx] = (half_t)gy;
    }
  }
}

// ---------------- host launcher ----------------

extern "C" void kernel_launch(void* const* d_in, const int* in_sizes, int n_in,
                              void* d_out, int out_size, void* d_ws, size_t ws_size,
                              hipStream_t stream) {
  (void)in_sizes; (void)n_in; (void)out_size; (void)ws_size;
  const float* x      = (const float*)d_in[0];
  const float* ln1_g  = (const float*)d_in[1];
  const float* ln1_b  = (const float*)d_in[2];
  const float* ln2_g  = (const float*)d_in[3];
  const float* ln2_b  = (const float*)d_in[4];
  const float* ln3_g  = (const float*)d_in[5];
  const float* ln3_b  = (const float*)d_in[6];
  const float* ln4_g  = (const float*)d_in[7];
  const float* ln4_b  = (const float*)d_in[8];
  const float* qkv1_b = (const float*)d_in[10];
  const float* out1_b = (const float*)d_in[12];
  const float* qkv2_b = (const float*)d_in[14];
  const float* out2_b = (const float*)d_in[16];
  const float* m1b1   = (const float*)d_in[18];
  const float* m1b2   = (const float*)d_in[20];
  const float* m2b1   = (const float*)d_in[22];
  const float* m2b2   = (const float*)d_in[24];

  char* ws = (char*)d_ws;
  size_t off = 0;
  auto take = [&](size_t bytes) -> char* {
    char* p = ws + off;
    off += (bytes + 255) & ~(size_t)255;
    return p;
  };
  const size_t E = (size_t)NTOK * D_;
  half_t* wbuf = (half_t*)take((size_t)393216 * 2); // all weights as f16
  half_t* A16  = (half_t*)take(E * 2);              // LN/gathered attn input
  float*  S32  = (float*)take(E * 4);               // attn output
  float*  R32  = (float*)take(E * 4);               // residual
  half_t* H16  = (half_t*)take(E * 2);              // LN output for MLP
  half_t* M16  = (half_t*)take((size_t)NTOK * 512 * 2); // MLP hidden
  float*  O32  = (float*)take(E * 4);               // branch-1 output

  half_t* wq1 = wbuf + 0;       // qkv1_w [384,128]
  half_t* wo1 = wbuf + 49152;   // out1_w [128,128]
  half_t* wq2 = wbuf + 65536;   // qkv2_w
  half_t* wo2 = wbuf + 114688;  // out2_w
  half_t* w11 = wbuf + 131072;  // mlp1_w1 [512,128]
  half_t* w12 = wbuf + 196608;  // mlp1_w2 [128,512]
  half_t* w21 = wbuf + 262144;  // mlp2_w1
  half_t* w22 = wbuf + 327680;  // mlp2_w2

  struct { int idx; size_t o; int n; } wc[8] = {
    {9, 0, 49152}, {11, 49152, 16384}, {13, 65536, 49152}, {15, 114688, 16384},
    {17, 131072, 65536}, {19, 196608, 65536}, {21, 262144, 65536}, {23, 327680, 65536}};
  for (int i = 0; i < 8; ++i)
    cast_w_kernel<<<(wc[i].n + 255) / 256, 256, 0, stream>>>(
        (const float*)d_in[wc[i].idx], wbuf + wc[i].o, wc[i].n);

  const int LN_GRID = NTOK / 8;           // 8 tokens (waves) per 256-thread block
  const int AT_GRID = NWTOT / 2;          // 2 windows (waves) per 64-thread block
  const int EW_GRID = (int)(E / 4 / 256);
  const int MT      = NTOK / 16;          // 8192 M-tiles
  const size_t SH1  = (size_t)1 * 16 * 128 * 2;   // fc1 LDS panel (N=512,K=128)
  const size_t SH2  = (size_t)4 * 16 * 512 * 2;   // fc2 LDS panel (N=128,K=512)

  // ---- Branch 1: W-MSA ----
  ln_gather1_kernel<<<LN_GRID, 256, 0, stream>>>(x, ln1_g, ln1_b, A16);
  attn_kernel<<<AT_GRID, 64, 0, stream>>>(A16, wq1, qkv1_b, wo1, out1_b, S32);
  res1_kernel<<<EW_GRID, 256, 0, stream>>>(x, S32, R32);
  ln_plain_kernel<<<LN_GRID, 256, 0, stream>>>(R32, ln2_g, ln2_b, H16);
  gemm_kernel<<<MT, 256, SH1, stream>>>(H16, w11, m1b1, nullptr, M16, nullptr, NTOK, 512, 128);
  gemm_kernel<<<MT / 4, 256, SH2, stream>>>(M16, w12, m1b2, R32, nullptr, O32, NTOK, 128, 512);

  // ---- Branch 2: SW-MSA ----
  ln_roll_gather_kernel<<<LN_GRID, 256, 0, stream>>>(O32, ln3_g, ln3_b, A16);
  attn_kernel<<<AT_GRID, 64, 0, stream>>>(A16, wq2, qkv2_b, wo2, out2_b, S32);
  res2_kernel<<<LN_GRID, 256, 0, stream>>>(O32, S32, R32);
  ln_plain_kernel<<<LN_GRID, 256, 0, stream>>>(R32, ln4_g, ln4_b, H16);
  gemm_kernel<<<MT, 256, SH1, stream>>>(H16, w21, m2b1, nullptr, M16, nullptr, NTOK, 512, 128);
  gemm_kernel<<<MT / 4, 256, SH2, stream>>>(M16, w22, m2b2, R32, nullptr, (float*)d_out, NTOK, 128, 512);
}
